// RNN_6631429505062
// MI455X (gfx1250) — matrix operand order repaired and checked by reading, not compiled, not run on hardware
//
#include <hip/hip_runtime.h>
#include <hip/hip_bf16.h>
#include <math.h>

// Persistent fused RNN for MI455X (gfx1250, wave32, WMMA).
//   h_{l+1} = tanh(x_l @ W_x^T + h_l @ W_h^T + b)
// 4 workgroups x 512 threads (16 waves). Each WG owns 16 batch rows; each wave
// owns one 16-wide N-tile of the hidden state. Weights are converted to bf16
// once and held in VGPRs as WMMA B-fragments for all 2048 steps. h and the
// current x tile live in double-buffered LDS (bf16); one s_barrier per step.
// tanh uses the gfx1250 V_TANH_F32 TRANS op (branch-free) instead of OCML.

typedef __bf16 bf16_t;
typedef __attribute__((ext_vector_type(16))) __bf16 v16bf;
typedef __attribute__((ext_vector_type(8)))  __bf16 v8bf;
typedef __attribute__((ext_vector_type(4)))  __bf16 v4bf;
typedef __attribute__((ext_vector_type(8)))  float  v8f;
typedef __attribute__((ext_vector_type(4)))  float  v4f;

#define BATCH  64
#define LENGTH 2048
#define INPUT  128
#define HIDDEN 256

#define M_TILE  16                 // batch rows per workgroup
#define NWAVES  16                 // one 16-col N tile per wave
#define THREADS (NWAVES * 32)

// Row strides (in bf16 elements) chosen so row-to-row bank shift = 8 banks:
// 32 lanes x 16B A-frag loads split into exactly two conflict-free passes.
#define HROW 272                   // 256 + 16  -> 544 B stride
#define XROW 144                   // 128 + 16  -> 288 B stride

union AFrag {
    struct { v8bf lo, hi; } h;
    v16bf v;
};

// Branch-free tanh on the critical serial path. gfx1250 has V_TANH_F32.
static __device__ __forceinline__ float fast_tanh(float v) {
#if __has_builtin(__builtin_amdgcn_tanhf)
    return __builtin_amdgcn_tanhf(v);
#elif __has_builtin(__builtin_amdgcn_tanh_f32)
    return __builtin_amdgcn_tanh_f32(v);
#else
    // tanh(x) = 1 - 2/(exp(2x)+1); exp2-based, saturates to +/-1, no branches.
    float e = __builtin_amdgcn_exp2f(v * 2.8853900817779268f); // 2*log2(e)*x
    return 1.0f - 2.0f * __builtin_amdgcn_rcpf(e + 1.0f);
#endif
}

__global__ __launch_bounds__(THREADS)
void rnn_persistent(const float* __restrict__ x,
                    const float* __restrict__ h0,
                    const float* __restrict__ Wx,
                    const float* __restrict__ Wh,
                    const float* __restrict__ bh,
                    float* __restrict__ out)
{
    __shared__ bf16_t hbuf[2][M_TILE][HROW];
    __shared__ bf16_t xbuf[2][M_TILE][XROW];

    const int tid  = threadIdx.x;
    const int lane = tid & 31;
    const int wave = tid >> 5;          // 0..15 -> N tile
    const int half = lane >> 4;         // 0/1: lane group
    const int l16  = lane & 15;
    const int b0   = blockIdx.x * M_TILE;
    const int n0   = wave * 16;

    // ---------------- time-invariant B fragments (weights) in VGPRs --------
    // y[b,n] = sum_k A[b,k] * W[n,k]   =>  B[k][n] = W[n][k]
    // B 32x16 layout: lane<16: n=n0+lane, holds K = s*32 + e      (e=0..15)
    //                 lane>=16: n=n0+lane-16, K = s*32 + 16 + e
    // -> 16 consecutive floats of row n of W: one 64B read per fragment/lane.
    const int nB = n0 + l16;

    v16bf bWh[HIDDEN / 32];
#pragma unroll
    for (int s = 0; s < HIDDEN / 32; ++s) {
        const float* src = Wh + (size_t)nB * HIDDEN + s * 32 + half * 16;
        v16bf f;
#pragma unroll
        for (int e = 0; e < 16; ++e) f[e] = (bf16_t)src[e];
        bWh[s] = f;
    }
    v16bf bWx[INPUT / 32];
#pragma unroll
    for (int s = 0; s < INPUT / 32; ++s) {
        const float* src = Wx + (size_t)nB * INPUT + s * 32 + half * 16;
        v16bf f;
#pragma unroll
        for (int e = 0; e < 16; ++e) f[e] = (bf16_t)src[e];
        bWx[s] = f;
    }
    const float bias = bh[nB];

    // ---------------- stage h(0) and x(0) into LDS as bf16 -----------------
    {
        const int m = tid >> 5;                   // 0..15
        const int c = (tid & 31) * 8;             // 8 h elems / thread
        const float* hs = h0 + (size_t)(b0 + m) * HIDDEN + c;
        v4f h4a = *(const v4f*)(hs);
        v4f h4b = *(const v4f*)(hs + 4);
#pragma unroll
        for (int e = 0; e < 4; ++e) {
            hbuf[0][m][c + e]     = (bf16_t)h4a[e];
            hbuf[0][m][c + 4 + e] = (bf16_t)h4b[e];
        }
        const int xc = (tid & 31) * 4;            // 4 x elems / thread
        v4f x4 = *(const v4f*)(x + ((size_t)(b0 + m) * LENGTH + 0) * INPUT + xc);
        v4bf xp;
#pragma unroll
        for (int e = 0; e < 4; ++e) xp[e] = (bf16_t)x4[e];
        *(v4bf*)&xbuf[0][m][xc] = xp;
    }
    __syncthreads();

    // ---------------- sequential scan over time ----------------------------
    const int xm = tid >> 5;
    const int xc = (tid & 31) * 4;

    for (int l = 0; l < LENGTH; ++l) {
        const int p = l & 1;

        // (1) prefetch next step's x tile into registers (overlaps WMMAs)
        v4f xr = {0.f, 0.f, 0.f, 0.f};
        if (l + 1 < LENGTH) {
            xr = *(const v4f*)(x + ((size_t)(b0 + xm) * LENGTH + (l + 1)) * INPUT + xc);
        }

        // (2) acc = bias; acc += h @ Wh^T; acc += x @ Wx^T   (f32 accum)
        v8f acc;
#pragma unroll
        for (int r = 0; r < 8; ++r) acc[r] = bias;

        const int m = l16;                         // A row for this lane
#pragma unroll
        for (int s = 0; s < HIDDEN / 32; ++s) {
            const int off = s * 32 + half * 16;    // A: K = off..off+15
            AFrag a;
            a.h.lo = *(const v8bf*)&hbuf[p][m][off];
            a.h.hi = *(const v8bf*)&hbuf[p][m][off + 8];
            acc = __builtin_amdgcn_wmma_f32_16x16x32_bf16(
                      false, a.v, false, bWh[s], (short)0, acc, false, false);
        }
#pragma unroll
        for (int s = 0; s < INPUT / 32; ++s) {
            const int off = s * 32 + half * 16;
            AFrag a;
            a.h.lo = *(const v8bf*)&xbuf[p][m][off];
            a.h.hi = *(const v8bf*)&xbuf[p][m][off + 8];
            acc = __builtin_amdgcn_wmma_f32_16x16x32_bf16(
                      false, a.v, false, bWx[s], (short)0, acc, false, false);
        }

        // (3) tanh, stream f32 result to out[l], stash bf16 h for next step.
        // D layout: VGPR r -> row (r + half*8), col = n0 + l16.
        float* orow = out + (size_t)l * BATCH * HIDDEN;
#pragma unroll
        for (int r = 0; r < 8; ++r) {
            const int mm = r + half * 8;
            const float hv = fast_tanh(acc[r]);
            orow[(size_t)(b0 + mm) * HIDDEN + n0 + l16] = hv;
            hbuf[1 - p][mm][n0 + l16] = (bf16_t)hv;
        }

        // (4) commit prefetched x into the other LDS buffer
        if (l + 1 < LENGTH) {
            v4bf xp;
#pragma unroll
            for (int e = 0; e < 4; ++e) xp[e] = (bf16_t)xr[e];
            *(v4bf*)&xbuf[1 - p][xm][xc] = xp;
        }

        __syncthreads();
    }
}

extern "C" void kernel_launch(void* const* d_in, const int* in_sizes, int n_in,
                              void* d_out, int out_size, void* d_ws, size_t ws_size,
                              hipStream_t stream) {
    (void)in_sizes; (void)n_in; (void)d_ws; (void)ws_size; (void)out_size;
    const float* x  = (const float*)d_in[0];
    const float* h0 = (const float*)d_in[1];
    const float* Wx = (const float*)d_in[2];
    const float* Wh = (const float*)d_in[3];
    const float* bh = (const float*)d_in[4];
    float* out = (float*)d_out;

    dim3 grid(BATCH / M_TILE);   // 4 persistent workgroups
    dim3 block(THREADS);         // 512 threads = 16 waves
    rnn_persistent<<<grid, block, 0, stream>>>(x, h0, Wx, Wh, bh, out);
}
